// PerformerEncoder_55164559950304
// MI455X (gfx1250) — compile-verified, matrix operand order checked by reading, not run on hardware
//
#include <hip/hip_runtime.h>

// ---------------- model constants ----------------
#define Dm   256        // hidden size
#define Hh   4          // heads
#define DHd  64         // dim per head
#define Ll   4          // layers
#define MV   266        // nb_features (valid)
#define MP   288        // nb_features padded (mult of 32)
#define Ss   4096       // seq len
#define Bb   4          // batch
#define BS   (Bb*Ss)    // 16384 rows
#define BH   (Bb*Hh)    // 16 (b,h) batches

#define DN_SCALE 0.35355339059327373f   // 64^-0.25
#define RATIO    0.06131393394849658f   // 266^-0.5
#define DIAGC    0.0625f                // 0.5 * 64^-0.5

typedef __bf16 bf16;
typedef __attribute__((ext_vector_type(16))) __bf16 v16bf;
typedef __attribute__((ext_vector_type(8)))  float  v8f;

// ---------------- bf16 WMMA GEMM (row-major A x pre-transposed B) ----------------
// C[m,n] = alpha * sum_k A[m*lda+k] * Bt[n*ldb+k]  + bias[n] (+ resid)
// Each wave: one 16(M) x 32(N) tile -> 1 A fragment, 2 B fragments, 2 WMMAs per K-step.
// act: 0=none 1=gelu(tanh approx) 2=tanh
struct GemmP {
    const bf16* A; const bf16* B; const float* bias; const float* resid;
    float* Cf; bf16* Cb;
    int M, N, K, lda, ldb, ldc, nb2;
    long long bsA1, bsA2, bsB1, bsB2, bsC1, bsC2;
    float alpha; int act;
};

__device__ __forceinline__ float apply_act(float v, int act) {
    if (act == 1) { float t = v; v = 0.5f*t*(1.f + tanhf(0.7978845608028654f*(t + 0.044715f*t*t*t))); }
    else if (act == 2) { v = tanhf(v); }
    return v;
}

__global__ void k_gemm(GemmP p) {
    const int lane = threadIdx.x & 31;
    const int wave = threadIdx.x >> 5;
    const int Mtiles = p.M >> 4;
    const int mt = blockIdx.x * 4 + wave;
    if (mt >= Mtiles) return;               // wave-uniform exit, EXEC stays all-1
    const int m0 = mt << 4;
    const int n0 = blockIdx.y << 5;         // 32-wide N tile
    const int z  = blockIdx.z;
    const long long offA = (long long)(z / p.nb2) * p.bsA1 + (long long)(z % p.nb2) * p.bsA2;
    const long long offB = (long long)(z / p.nb2) * p.bsB1 + (long long)(z % p.nb2) * p.bsB2;
    const long long offC = (long long)(z / p.nb2) * p.bsC1 + (long long)(z % p.nb2) * p.bsC2;
    const int half = lane >> 4;
    const int r16  = lane & 15;

    const bf16* pa0 = p.A + offA + (size_t)(m0 + r16) * p.lda;
    const bf16* pb0 = p.B + offB + (size_t)(n0 + r16) * p.ldb;
    const bf16* pb1 = pb0 + (size_t)16 * p.ldb;

    v8f acc0 = {};
    v8f acc1 = {};
    for (int k0 = 0; k0 < p.K; k0 += 32) {
        v16bf af, bf0, bf1;
#pragma unroll
        for (int i = 0; i < 8; ++i) {
            int kb = k0 + ((i < 4) ? (2 * i) : (16 + 2 * (i - 4))) + half * 8;
            af[2*i]  = pa0[kb];  af[2*i+1]  = pa0[kb+1];
            bf0[2*i] = pb0[kb];  bf0[2*i+1] = pb0[kb+1];
            bf1[2*i] = pb1[kb];  bf1[2*i+1] = pb1[kb+1];
        }
        if (k0 + 32 < p.K) {
            __builtin_prefetch((const void*)(pa0 + k0 + 64), 0, 1);
            __builtin_prefetch((const void*)(pb0 + k0 + 64), 0, 1);
            __builtin_prefetch((const void*)(pb1 + k0 + 64), 0, 1);
        }
        acc0 = __builtin_amdgcn_wmma_f32_16x16x32_bf16(
                   false, af, false, bf0, (short)0, acc0, false, false);
        acc1 = __builtin_amdgcn_wmma_f32_16x16x32_bf16(
                   false, af, false, bf1, (short)0, acc1, false, false);
    }

    const int nA = n0 + r16;
    const int nB = n0 + 16 + r16;
    const float bvA = p.bias ? p.bias[nA] : 0.f;
    const float bvB = p.bias ? p.bias[nB] : 0.f;
#pragma unroll
    for (int r = 0; r < 8; ++r) {
        int m = m0 + half * 8 + r;
        long long ciA = offC + (long long)m * p.ldc + nA;
        long long ciB = offC + (long long)m * p.ldc + nB;
        float vA = p.alpha * acc0[r] + bvA;
        float vB = p.alpha * acc1[r] + bvB;
        if (p.resid) { vA += p.resid[ciA]; vB += p.resid[ciB]; }
        vA = apply_act(vA, p.act);
        vB = apply_act(vB, p.act);
        if (p.Cf) { p.Cf[ciA] = vA; p.Cf[ciB] = vB; }
        if (p.Cb) { p.Cb[ciA] = (bf16)vA; p.Cb[ciB] = (bf16)vB; }
    }
}

// ---------------- layernorm (one wave per row, D=256) ----------------
__global__ void k_layernorm(const float* __restrict__ x, const float* __restrict__ addpos,
                            const float* __restrict__ g, const float* __restrict__ b,
                            float* outF, bf16* outB, int rows) {
    int wid = threadIdx.x >> 5, lane = threadIdx.x & 31;
    int row = blockIdx.x * (blockDim.x >> 5) + wid;
    if (row >= rows) return;
    const float* xr = x + (size_t)row * Dm;
    const float* pr = addpos ? addpos + (size_t)(row % Ss) * Dm : nullptr;
    float v[8]; float sum = 0.f, sq = 0.f;
#pragma unroll
    for (int i = 0; i < 8; ++i) {
        int c = lane + 32 * i;
        float t = xr[c] + (pr ? pr[c] : 0.f);
        v[i] = t; sum += t; sq += t * t;
    }
    for (int o = 16; o > 0; o >>= 1) { sum += __shfl_xor(sum, o, 32); sq += __shfl_xor(sq, o, 32); }
    float mu = sum * (1.f / Dm);
    float var = sq * (1.f / Dm) - mu * mu;
    float rstd = rsqrtf(var + 1e-12f);
#pragma unroll
    for (int i = 0; i < 8; ++i) {
        int c = lane + 32 * i;
        float t = (v[i] - mu) * rstd * g[c] + b[c];
        if (outF) outF[(size_t)row * Dm + c] = t;
        if (outB) outB[(size_t)row * Dm + c] = (bf16)t;
    }
}

// ---------------- converts ----------------
__global__ void k_cvt_transpose(const float* __restrict__ in, bf16* __restrict__ out, int K, int N) {
    int idx = blockIdx.x * blockDim.x + threadIdx.x;
    if (idx >= K * N) return;
    int n = idx / K, k = idx - n * K;
    out[idx] = (bf16)in[(size_t)k * N + n];
}

__global__ void k_cvt_pad_rows(const float* __restrict__ in, bf16* __restrict__ out,
                               int rows_in, int rows_out, int cols) {
    int idx = blockIdx.x * blockDim.x + threadIdx.x;
    if (idx >= rows_out * cols) return;
    int r = idx / cols, c = idx - r * cols;
    out[idx] = (r < rows_in) ? (bf16)in[(size_t)r * cols + c] : (bf16)0.f;
}

// v [BS, Dm] (heads packed) -> vT [bh][DHd][Ss]
__global__ void k_vT(const bf16* __restrict__ v, bf16* __restrict__ vT) {
    int idx = blockIdx.x * blockDim.x + threadIdx.x;
    if (idx >= BH * Ss * DHd) return;
    int bh = idx / (Ss * DHd); int rem = idx - bh * Ss * DHd;
    int s = rem / DHd; int dd = rem - s * DHd;
    int b = bh / Hh, h = bh - b * Hh;
    vT[((size_t)bh * DHd + dd) * Ss + s] = v[((size_t)(b * Ss + s)) * Dm + h * DHd + dd];
}

// ---------------- FAVOR+ feature map ----------------
// row = (b*H+h)*S + s ; blockDim = MP
// gstab==null -> per-row stabilizer (queries) ; kt!=0 -> write transposed [bh][MP][Ss]
__global__ void k_phi(const float* __restrict__ dash, const float* __restrict__ qk,
                      const float* __restrict__ gstab, bf16* __restrict__ out, int kt) {
    int row = blockIdx.x;
    int bh = row / Ss; int s = row - bh * Ss; int b = bh / Hh; int h = bh - b * Hh;
    int m = threadIdx.x;
    __shared__ float red[9];
    __shared__ float sh2[2];
    const float* dr = dash + (size_t)row * MP;
    // diag = 0.5 * dn^2 * sum_d qk^2
    float pp = 0.f;
    if (m < DHd) { float qv = qk[((size_t)(b * Ss + s)) * Dm + h * DHd + m]; pp = qv * qv; }
    float ws = pp;
    for (int o = 16; o > 0; o >>= 1) ws += __shfl_xor(ws, o, 32);
    int wid = m >> 5;
    if ((m & 31) == 0) red[wid] = ws;
    __syncthreads();
    if (m == 0) { float su = 0.f; for (int i = 0; i < 9; ++i) su += red[i]; sh2[0] = su * DIAGC; }
    __syncthreads();
    // stabilizer
    float val = (m < MV) ? dr[m] : -3.0e38f;
    float wm = val;
    for (int o = 16; o > 0; o >>= 1) wm = fmaxf(wm, __shfl_xor(wm, o, 32));
    if ((m & 31) == 0) red[wid] = wm;
    __syncthreads();
    if (m == 0) {
        float mx = -3.0e38f; for (int i = 0; i < 9; ++i) mx = fmaxf(mx, red[i]);
        sh2[1] = gstab ? *gstab : mx;
    }
    __syncthreads();
    float o_ = 0.f;
    if (m < MV) o_ = RATIO * (expf(val - sh2[0] - sh2[1]) + 1e-4f);
    size_t oidx = kt ? (((size_t)bh * MP + m) * (size_t)Ss + s)
                     : ((size_t)row * MP + m);
    out[oidx] = (bf16)o_;
}

// ---------------- global max (keys) ----------------
__global__ void k_max1(const float* __restrict__ dash, float* __restrict__ bmax) {
    __shared__ float sm[256];
    long long total = (long long)BH * Ss * MV;
    long long stride = (long long)gridDim.x * blockDim.x;
    float mx = -3.0e38f;
    for (long long i = (long long)blockIdx.x * blockDim.x + threadIdx.x; i < total; i += stride) {
        long long row = i / MV; int m = (int)(i - row * MV);
        mx = fmaxf(mx, dash[row * MP + m]);
    }
    sm[threadIdx.x] = mx; __syncthreads();
    for (int s2 = 128; s2 > 0; s2 >>= 1) {
        if (threadIdx.x < s2) sm[threadIdx.x] = fmaxf(sm[threadIdx.x], sm[threadIdx.x + s2]);
        __syncthreads();
    }
    if (threadIdx.x == 0) bmax[blockIdx.x] = sm[0];
}
__global__ void k_max2(const float* __restrict__ bmax, float* __restrict__ g) {
    __shared__ float sm[1024];
    sm[threadIdx.x] = bmax[threadIdx.x]; __syncthreads();
    for (int s2 = 512; s2 > 0; s2 >>= 1) {
        if (threadIdx.x < s2) sm[threadIdx.x] = fmaxf(sm[threadIdx.x], sm[threadIdx.x + s2]);
        __syncthreads();
    }
    if (threadIdx.x == 0) g[0] = sm[0];
}

// ---------------- ksum / d_inv ----------------
// kpT [bh][MP][Ss] -> ksum[bh][m] (contiguous reduction)
__global__ void k_ksum(const bf16* __restrict__ kpT, float* __restrict__ ksum) {
    int idx = blockIdx.x * blockDim.x + threadIdx.x;
    if (idx >= BH * MP) return;
    const bf16* base = kpT + (size_t)idx * Ss;
    float s = 0.f;
    for (int t = 0; t < Ss; ++t) s += (float)base[t];
    ksum[idx] = s;
}
__global__ void k_dinv(const bf16* __restrict__ qp, const float* __restrict__ ksum,
                       float* __restrict__ dinv) {
    int wid = threadIdx.x >> 5, lane = threadIdx.x & 31;
    int row = blockIdx.x * (blockDim.x >> 5) + wid;
    if (row >= BH * Ss) return;
    int bh = row / Ss;
    const bf16* qr = qp + (size_t)row * MP;
    const float* ks = ksum + (size_t)bh * MP;
    float a = 0.f;
    for (int m = lane; m < MP; m += 32) a += (float)qr[m] * ks[m];
    for (int o = 16; o > 0; o >>= 1) a += __shfl_xor(a, o, 32);
    if (lane == 0) dinv[row] = 1.f / a;
}

// ---------------- small reshuffles ----------------
__global__ void k_ctxT(const float* __restrict__ ctx, bf16* __restrict__ ctxT) {
    int idx = blockIdx.x * blockDim.x + threadIdx.x;
    if (idx >= BH * MP * DHd) return;
    int bh = idx / (MP * DHd); int rem = idx - bh * MP * DHd;
    int m = rem / DHd; int dd = rem - m * DHd;
    ctxT[(size_t)bh * DHd * MP + (size_t)dd * MP + m] = (bf16)ctx[idx];
}
__global__ void k_attn(const float* __restrict__ outs, const float* __restrict__ dinv,
                       bf16* __restrict__ attn) {
    int idx = blockIdx.x * blockDim.x + threadIdx.x;
    if (idx >= BH * Ss * DHd) return;
    int bh = idx / (Ss * DHd); int rem = idx - bh * Ss * DHd;
    int s = rem / DHd; int dd = rem - s * DHd;
    int b = bh / Hh, h = bh - b * Hh;
    attn[((size_t)(b * Ss + s)) * Dm + h * DHd + dd] = (bf16)(outs[idx] * dinv[(size_t)bh * Ss + s]);
}

// ---------------- attention pooling ----------------
__global__ void k_alpha(const bf16* __restrict__ e, const float* __restrict__ p2w,
                        const float* __restrict__ p2b, const float* __restrict__ mask,
                        float* __restrict__ alpha) {
    int wid = threadIdx.x >> 5, lane = threadIdx.x & 31;
    int row = blockIdx.x * (blockDim.x >> 5) + wid;
    if (row >= BS) return;
    const bf16* er = e + (size_t)row * Dm;
    float a = 0.f;
    for (int c = lane; c < Dm; c += 32) a += (float)er[c] * p2w[c];
    for (int o = 16; o > 0; o >>= 1) a += __shfl_xor(a, o, 32);
    if (lane == 0) alpha[row] = expf(a + p2b[0]) * mask[row];
}
__global__ void k_alpha_sum(const float* __restrict__ alpha, float* __restrict__ asum) {
    __shared__ float sm[256];
    int b = blockIdx.x; float s = 0.f;
    for (int i = threadIdx.x; i < Ss; i += 256) s += alpha[(size_t)b * Ss + i];
    sm[threadIdx.x] = s; __syncthreads();
    for (int s2 = 128; s2 > 0; s2 >>= 1) {
        if (threadIdx.x < s2) sm[threadIdx.x] += sm[threadIdx.x + s2];
        __syncthreads();
    }
    if (threadIdx.x == 0) asum[b] = sm[0];
}
__global__ void k_pool(const float* __restrict__ x, const float* __restrict__ alpha,
                       const float* __restrict__ asum, float* __restrict__ out) {
    int b = blockIdx.x; int d = threadIdx.x;
    float inv = 1.f / (asum[b] + 1e-8f);
    float acc = 0.f;
    for (int s = 0; s < Ss; ++s)
        acc += x[((size_t)b * Ss + s) * Dm + d] * alpha[(size_t)b * Ss + s];
    out[(size_t)b * Dm + d] = acc * inv;
}

// ---------------- host side ----------------
static void gemm(hipStream_t st, const bf16* A, const bf16* B, const float* bias,
                 const float* resid, float* Cf, bf16* Cb,
                 int M, int N, int K, int lda, int ldb, int ldc,
                 int nbatch, int nb2,
                 long long bsA1, long long bsA2, long long bsB1, long long bsB2,
                 long long bsC1, long long bsC2, float alpha, int act) {
    GemmP p{A, B, bias, resid, Cf, Cb, M, N, K, lda, ldb, ldc, nb2,
            bsA1, bsA2, bsB1, bsB2, bsC1, bsC2, alpha, act};
    dim3 grid((unsigned)((M / 16 + 3) / 4), (unsigned)(N / 32), (unsigned)nbatch);
    k_gemm<<<grid, 128, 0, st>>>(p);
}

extern "C" void kernel_launch(void* const* d_in, const int* in_sizes, int n_in,
                              void* d_out, int out_size, void* d_ws, size_t ws_size,
                              hipStream_t stream) {
    (void)in_sizes; (void)n_in; (void)out_size; (void)ws_size;
    const float* in_embs = (const float*)d_in[0];
    const float* mask    = (const float*)d_in[1];
    const float* pos     = (const float*)d_in[2];
    const float* ln_g    = (const float*)d_in[3];
    const float* ln_b    = (const float*)d_in[4];
    const float* proj    = (const float*)d_in[5];
    const float* Wq = (const float*)d_in[6];   const float* bq = (const float*)d_in[7];
    const float* Wk = (const float*)d_in[8];   const float* bk = (const float*)d_in[9];
    const float* Wv = (const float*)d_in[10];  const float* bv = (const float*)d_in[11];
    const float* Wo = (const float*)d_in[12];  const float* bo = (const float*)d_in[13];
    const float* ln1_g = (const float*)d_in[14]; const float* ln1_b = (const float*)d_in[15];
    const float* W1 = (const float*)d_in[16];  const float* b1 = (const float*)d_in[17];
    const float* W2 = (const float*)d_in[18];  const float* b2 = (const float*)d_in[19];
    const float* ln2_g = (const float*)d_in[20]; const float* ln2_b = (const float*)d_in[21];
    const float* p1w = (const float*)d_in[22]; const float* p1b = (const float*)d_in[23];
    const float* p2w = (const float*)d_in[24]; const float* p2b = (const float*)d_in[25];

    // workspace carve-out
    char* w = (char*)d_ws; size_t off = 0;
    auto alloc = [&](size_t bytes) -> void* {
        void* p = w + off; off = (off + bytes + 255) & ~(size_t)255; return p;
    };
    float* x      = (float*)alloc((size_t)BS * Dm * 4);
    bf16*  xn_bf  = (bf16*) alloc((size_t)BS * Dm * 2);
    float* qf     = (float*)alloc((size_t)BS * Dm * 4);
    float* kf     = (float*)alloc((size_t)BS * Dm * 4);
    bf16*  qbf    = (bf16*) alloc((size_t)BS * Dm * 2);
    bf16*  kbf    = (bf16*) alloc((size_t)BS * Dm * 2);
    bf16*  vbf    = (bf16*) alloc((size_t)BS * Dm * 2);
    bf16*  vT     = (bf16*) alloc((size_t)BH * DHd * Ss * 2);
    float* dash   = (float*)alloc((size_t)BH * Ss * MP * 4);   // reused as FFN hidden & pooling-e
    bf16*  qp     = (bf16*) alloc((size_t)BH * Ss * MP * 2);   // row-major [row][MP]
    bf16*  kpT    = (bf16*) alloc((size_t)BH * MP * Ss * 2);   // transposed [bh][MP][Ss]
    float* ksumb  = (float*)alloc((size_t)BH * MP * 4);
    float* bmax   = (float*)alloc(1024 * 4);
    float* gmax   = (float*)alloc(4);
    float* ctx    = (float*)alloc((size_t)BH * MP * DHd * 4);
    bf16*  ctxT   = (bf16*) alloc((size_t)BH * DHd * MP * 2);
    float* dinv   = (float*)alloc((size_t)BH * Ss * 4);
    float* alpha  = (float*)alloc((size_t)BS * 4);
    float* asum   = (float*)alloc(Bb * 4);
    bf16*  wqt    = (bf16*) alloc((size_t)Dm * Dm * 2);
    bf16*  wkt    = (bf16*) alloc((size_t)Dm * Dm * 2);
    bf16*  wvt    = (bf16*) alloc((size_t)Dm * Dm * 2);
    bf16*  wot    = (bf16*) alloc((size_t)Dm * Dm * 2);
    bf16*  w1t    = (bf16*) alloc((size_t)Dm * 4 * Dm * 2);
    bf16*  w2t    = (bf16*) alloc((size_t)Dm * 4 * Dm * 2);
    bf16*  projb  = (bf16*) alloc((size_t)MP * DHd * 2);
    bf16*  p1t    = (bf16*) alloc((size_t)Dm * Dm * 2);
    // aliases into dead buffers
    float* outs    = qf;            // q fp32 dead after phi(q)
    bf16*  attn_bf = (bf16*)kf;     // k fp32 dead after phi(k)
    bf16*  h1      = (bf16*)dash;   // dash dead once kp is built
    bf16*  e_bf    = (bf16*)dash;   // pooling-e after all layers

    // x = LN(input + pos)
    k_layernorm<<<BS / 8, 256, 0, stream>>>(in_embs, pos, ln_g, ln_b, x, nullptr, BS);

    for (int l = 0; l < Ll; ++l) {
        const float* Wq_l = Wq + (size_t)l * Dm * Dm;  const float* bq_l = bq + (size_t)l * Dm;
        const float* Wk_l = Wk + (size_t)l * Dm * Dm;  const float* bk_l = bk + (size_t)l * Dm;
        const float* Wv_l = Wv + (size_t)l * Dm * Dm;  const float* bv_l = bv + (size_t)l * Dm;
        const float* Wo_l = Wo + (size_t)l * Dm * Dm;  const float* bo_l = bo + (size_t)l * Dm;
        const float* W1_l = W1 + (size_t)l * Dm * 4 * Dm; const float* b1_l = b1 + (size_t)l * 4 * Dm;
        const float* W2_l = W2 + (size_t)l * Dm * 4 * Dm; const float* b2_l = b2 + (size_t)l * Dm;
        const float* proj_l = proj + (size_t)l * MV * DHd;

        // weight conversions (deterministic each call)
        int nW = Dm * Dm;
        k_cvt_transpose<<<(nW + 255) / 256, 256, 0, stream>>>(Wq_l, wqt, Dm, Dm);
        k_cvt_transpose<<<(nW + 255) / 256, 256, 0, stream>>>(Wk_l, wkt, Dm, Dm);
        k_cvt_transpose<<<(nW + 255) / 256, 256, 0, stream>>>(Wv_l, wvt, Dm, Dm);
        k_cvt_transpose<<<(nW + 255) / 256, 256, 0, stream>>>(Wo_l, wot, Dm, Dm);
        k_cvt_transpose<<<(Dm * 4 * Dm + 255) / 256, 256, 0, stream>>>(W1_l, w1t, Dm, 4 * Dm);
        k_cvt_transpose<<<(Dm * 4 * Dm + 255) / 256, 256, 0, stream>>>(W2_l, w2t, 4 * Dm, Dm);
        k_cvt_pad_rows<<<(MP * DHd + 255) / 256, 256, 0, stream>>>(proj_l, projb, MV, MP, DHd);

        // LN1 -> bf16
        k_layernorm<<<BS / 8, 256, 0, stream>>>(x, nullptr, ln1_g + (size_t)l * Dm,
                                                ln1_b + (size_t)l * Dm, nullptr, xn_bf, BS);
        // Q,K,V projections
        gemm(stream, xn_bf, wqt, bq_l, nullptr, qf, qbf, BS, Dm, Dm, Dm, Dm, Dm,
             1, 1, 0, 0, 0, 0, 0, 0, 1.f, 0);
        gemm(stream, xn_bf, wkt, bk_l, nullptr, kf, kbf, BS, Dm, Dm, Dm, Dm, Dm,
             1, 1, 0, 0, 0, 0, 0, 0, 1.f, 0);
        gemm(stream, xn_bf, wvt, bv_l, nullptr, nullptr, vbf, BS, Dm, Dm, Dm, Dm, Dm,
             1, 1, 0, 0, 0, 0, 0, 0, 1.f, 0);
        k_vT<<<(BH * Ss * DHd + 255) / 256, 256, 0, stream>>>(vbf, vT);

        // dash_q = dn * q . proj^T   (batched over 16 (b,h))
        gemm(stream, qbf, projb, nullptr, nullptr, dash, nullptr, Ss, MP, DHd, Dm, DHd, MP,
             BH, Hh, (long long)Ss * Dm, DHd, 0, 0,
             (long long)Hh * Ss * MP, (long long)Ss * MP, DN_SCALE, 0);
        k_phi<<<BH * Ss, MP, 0, stream>>>(dash, qf, nullptr, qp, 0);

        // dash_k + global stabilizer
        gemm(stream, kbf, projb, nullptr, nullptr, dash, nullptr, Ss, MP, DHd, Dm, DHd, MP,
             BH, Hh, (long long)Ss * Dm, DHd, 0, 0,
             (long long)Hh * Ss * MP, (long long)Ss * MP, DN_SCALE, 0);
        k_max1<<<1024, 256, 0, stream>>>(dash, bmax);
        k_max2<<<1, 1024, 0, stream>>>(bmax, gmax);
        k_phi<<<BH * Ss, MP, 0, stream>>>(dash, kf, gmax, kpT, 1);

        k_ksum<<<(BH * MP + 255) / 256, 256, 0, stream>>>(kpT, ksumb);

        // ctx[m,d] = sum_s kpT[m,s] * vT[d,s]   (pure row-major WMMA GEMM, K=4096)
        gemm(stream, kpT, vT, nullptr, nullptr, ctx, nullptr, MP, DHd, Ss, Ss, Ss, DHd,
             BH, Hh, (long long)Hh * MP * Ss, (long long)MP * Ss,
             (long long)Hh * DHd * Ss, (long long)DHd * Ss,
             (long long)Hh * MP * DHd, (long long)MP * DHd, 1.f, 0);
        k_ctxT<<<(BH * MP * DHd + 255) / 256, 256, 0, stream>>>(ctx, ctxT);

        // outs = qp @ ctx
        gemm(stream, qp, ctxT, nullptr, nullptr, outs, nullptr, Ss, DHd, MP, MP, MP, DHd,
             BH, Hh, (long long)Hh * Ss * MP, (long long)Ss * MP,
             (long long)Hh * DHd * MP, (long long)DHd * MP,
             (long long)Hh * Ss * DHd, (long long)Ss * DHd, 1.f, 0);

        k_dinv<<<BH * Ss / 8, 256, 0, stream>>>(qp, ksumb, dinv);
        k_attn<<<(BH * Ss * DHd + 255) / 256, 256, 0, stream>>>(outs, dinv, attn_bf);

        // x += attn @ Wo + bo
        gemm(stream, attn_bf, wot, bo_l, x, x, nullptr, BS, Dm, Dm, Dm, Dm, Dm,
             1, 1, 0, 0, 0, 0, 0, 0, 1.f, 0);

        // FFN
        k_layernorm<<<BS / 8, 256, 0, stream>>>(x, nullptr, ln2_g + (size_t)l * Dm,
                                                ln2_b + (size_t)l * Dm, nullptr, xn_bf, BS);
        gemm(stream, xn_bf, w1t, b1_l, nullptr, nullptr, h1, BS, 4 * Dm, Dm, Dm, Dm, 4 * Dm,
             1, 1, 0, 0, 0, 0, 0, 0, 1.f, 1 /*gelu*/);
        gemm(stream, h1, w2t, b2_l, x, x, nullptr, BS, Dm, 4 * Dm, 4 * Dm, 4 * Dm, Dm,
             1, 1, 0, 0, 0, 0, 0, 0, 1.f, 0);
    }

    // attention pooling
    k_cvt_transpose<<<(Dm * Dm + 255) / 256, 256, 0, stream>>>(p1w, p1t, Dm, Dm);
    k_cvt_pad_rows<<<(BS * Dm + 255) / 256, 256, 0, stream>>>(x, xn_bf, BS, BS, Dm);
    gemm(stream, xn_bf, p1t, p1b, nullptr, nullptr, e_bf, BS, Dm, Dm, Dm, Dm, Dm,
         1, 1, 0, 0, 0, 0, 0, 0, 1.f, 2 /*tanh*/);
    k_alpha<<<BS / 8, 256, 0, stream>>>(e_bf, p2w, p2b, mask, alpha);
    k_alpha_sum<<<Bb, 256, 0, stream>>>(alpha, asum);
    k_pool<<<Bb, Dm, 0, stream>>>(x, alpha, asum, (float*)d_out);
}
